// SpaceOpt_31688268710087
// MI455X (gfx1250) — matrix operand order repaired and checked by reading, not compiled, not run on hardware
//
#include <hip/hip_runtime.h>

typedef __attribute__((ext_vector_type(16))) _Float16 v16h;
typedef __attribute__((ext_vector_type(8)))  _Float16 v8h;
typedef __attribute__((ext_vector_type(4)))  _Float16 v4h;
typedef __attribute__((ext_vector_type(8)))  float    v8f;
typedef __attribute__((ext_vector_type(4)))  float    v4f;
typedef __attribute__((ext_vector_type(4)))  int      v4i;

#define WMMA_F16(a, b, c) \
  __builtin_amdgcn_wmma_f32_16x16x32_f16(false, (a), false, (b), (short)0, (c), false, false)

constexpr int Bn = 131072;
constexpr int Nn = 1024;
constexpr int Kn = 64;

// ---- CDNA5 async global->LDS copy path (ASYNCcnt-tracked) ----
#if __has_builtin(__builtin_amdgcn_global_load_async_to_lds_b128)
#define HAVE_ASYNC_LDS 1
static __device__ __forceinline__ void async_cp16(const void* g, void* l) {
  // signature (probe-confirmed): 4 args, v4i* pointers, imm offset, imm cpol
  __builtin_amdgcn_global_load_async_to_lds_b128((v4i*)g, (v4i*)l, 0, 0);
}
static __device__ __forceinline__ void async_wait0() {
#if __has_builtin(__builtin_amdgcn_s_wait_asynccnt)
  __builtin_amdgcn_s_wait_asynccnt(0);
#else
  asm volatile("s_wait_asynccnt 0" ::: "memory");
#endif
}
#else
#define HAVE_ASYNC_LDS 0
#endif

// Combine two contiguous v8h (16B each) into one WMMA v16h operand.
static __device__ __forceinline__ v16h frag2(const _Float16* p0, const _Float16* p1) {
  v8h a = *(const v8h*)p0;
  v8h b = *(const v8h*)p1;
  return __builtin_shufflevector(a, b, 0, 1, 2, 3, 4, 5, 6, 7,
                                       8, 9, 10, 11, 12, 13, 14, 15);
}

// ---- fragment-order offsets (B operand k order chosen here, same as A: lane = half*16+r,
//      element e -> K = half*16+e; compact 16 halfs per lane in global memory) ----

// Phase-1 B = T1 (N x K): chunk c = n>>5 (K=32 step), tile t = j>>4
static __device__ __forceinline__ size_t t1f_off(int n, int j) {
  const int c = n >> 5, k = n & 31;
  const int half = k >> 4, e = k & 15;
  const int t = j >> 4, rr2 = j & 15;
  const int l = half * 16 + rr2;
  return ((size_t)(c * 4 + t) * 32 + l) * 16 + e;
}

// Phase-2 B = wn (K x N): kstep s = k>>5, col tile ct = n>>4
static __device__ __forceinline__ size_t wnf_off(int k, int n) {
  const int s = k >> 5, kk = k & 31;
  const int half = kk >> 4, e = kk & 15;
  const int ct = n >> 4, rr2 = n & 15;
  const int l = half * 16 + rr2;
  return ((size_t)(ct * 2 + s) * 32 + l) * 16 + e;
}

// ---------------------------------------------------------------------------
// Kernel 1: wn = rownorm(r@w); Minv = phi^-1 (Gauss-Jordan, SPD, diag~1);
// T1 = wn^T @ Minv. Emits hi/lo f16 splits of T1 and wn, pre-swizzled into
// WMMA B-fragment order. One block; negligible next to the batch GEMMs.
// ---------------------------------------------------------------------------
__global__ __launch_bounds__(256) void spaceopt_prep(
    const float* __restrict__ w, const float* __restrict__ r,
    float* __restrict__ wn32,
    _Float16* __restrict__ t1f_hi, _Float16* __restrict__ t1f_lo,
    _Float16* __restrict__ wnf_hi, _Float16* __restrict__ wnf_lo)
{
  __shared__ float rs[64][64];
  __shared__ float P[64][65];
  __shared__ float Iv[64][65];
  __shared__ float norms[64];
  __shared__ float colv[64];

  const int tid = threadIdx.x;

  for (int i = tid; i < 64 * 64; i += 256) rs[i >> 6][i & 63] = r[i];
  if (tid < 64) norms[tid] = 0.f;
  __syncthreads();

  // w_new = r @ w, plus row sum-of-squares. 4 threads per output row.
  {
    const int i  = tid >> 2;
    const int n0 = (tid & 3) * 256;
    float ss = 0.f;
    for (int n = n0; n < n0 + 256; ++n) {
      float acc = 0.f;
      for (int k = 0; k < 64; ++k) acc += rs[i][k] * w[k * Nn + n];
      wn32[i * Nn + n] = acc;
      ss += acc * acc;
    }
    atomicAdd(&norms[i], ss);
  }
  __syncthreads();

  // normalize rows; emit hi/lo f16 split of wn in phase-2 fragment order
  {
    const int i  = tid >> 2;
    const int n0 = (tid & 3) * 256;
    const float sc = rsqrtf(norms[i]);
    for (int n = n0; n < n0 + 256; ++n) {
      float v = wn32[i * Nn + n] * sc;
      wn32[i * Nn + n] = v;
      _Float16 h = (_Float16)v;
      const size_t o = wnf_off(i, n);
      wnf_hi[o] = h;
      wnf_lo[o] = (_Float16)(v - (float)h);
    }
  }
  __syncthreads();

  // phi = wn @ wn^T (fp32), identity alongside
  for (int p = tid; p < 64 * 64; p += 256) {
    const int i = p >> 6, j = p & 63;
    float acc = 0.f;
    for (int n = 0; n < Nn; ++n) acc += wn32[i * Nn + n] * wn32[j * Nn + n];
    P[i][j]  = acc;
    Iv[i][j] = (i == j) ? 1.f : 0.f;
  }
  __syncthreads();

  // Gauss-Jordan inversion (phi is SPD with unit diagonal; no pivoting)
  for (int p = 0; p < 64; ++p) {
    if (tid < 64) colv[tid] = P[tid][p];
    __syncthreads();
    const float ip = 1.f / colv[p];
    if (tid < 64) { P[p][tid] *= ip; Iv[p][tid] *= ip; }
    __syncthreads();
    for (int q = tid; q < 64 * 64; q += 256) {
      const int ii = q >> 6, j = q & 63;
      if (ii != p) {
        const float f = colv[ii];
        P[ii][j]  -= f * P[p][j];
        Iv[ii][j] -= f * Iv[p][j];
      }
    }
    __syncthreads();
  }

  // T1 = wn^T @ Minv  (N x K), hi/lo split in phase-1 fragment order
  for (int n = tid; n < Nn; n += 256) {
    for (int jb = 0; jb < 64; jb += 8) {
      float acc[8] = {};
      for (int k = 0; k < 64; ++k) {
        const float vv = wn32[k * Nn + n];
        #pragma unroll
        for (int j = 0; j < 8; ++j) acc[j] += vv * Iv[k][jb + j];
      }
      #pragma unroll
      for (int j = 0; j < 8; ++j) {
        const float vv = acc[j];
        _Float16 h = (_Float16)vv;
        const size_t o = t1f_off(n, jb + j);
        t1f_hi[o] = h;
        t1f_lo[o] = (_Float16)(vv - (float)h);
      }
    }
  }
}

// ---------------------------------------------------------------------------
// Kernel 2: fused  v = x @ T1 ; proj = v @ wn  with split-f16 WMMA.
// 128 rows per block, 8 waves; each wave owns 16 rows.
// B operands arrive pre-swizzled; LDS copies & fragment reads are all b128
// (async global->LDS when available). Streaming x / proj / v use NT policy
// so the L2-resident T1/wn fragments (reused by all 1024 blocks) stay hot.
// LDS fragment per-lane stride = 24 halfs (48B) -> conflict-free bank pattern.
// ---------------------------------------------------------------------------
__global__ __launch_bounds__(256) void spaceopt_main(
    const float* __restrict__ x,
    const _Float16* __restrict__ t1f_hi, const _Float16* __restrict__ t1f_lo,
    const _Float16* __restrict__ wnf_hi, const _Float16* __restrict__ wnf_lo,
    float* __restrict__ proj, float* __restrict__ vout)
{
  __shared__ __align__(16) _Float16 Xh[128][40];   // 32-wide x chunk (hi), padded
  __shared__ __align__(16) _Float16 Xl[128][40];   // (lo)
  __shared__ __align__(16) _Float16 TfH[4][768];   // T1 fragments: [tile][lane*24+e]
  __shared__ __align__(16) _Float16 TfL[4][768];
  __shared__ __align__(16) _Float16 Vh[128][72];   // v tile (hi) for phase 2
  __shared__ __align__(16) _Float16 Vl[128][72];
  __shared__ __align__(16) _Float16 WfH[16][768];  // wn fragments: [tile*2+ks][lane*24+e]
  __shared__ __align__(16) _Float16 WfL[16][768];

  const int tid  = threadIdx.x;
  const int wv   = tid >> 5;
  const int lane = tid & 31;
  const int half = lane >> 4;
  const int rr   = lane & 15;
  const size_t m0 = (size_t)blockIdx.x * 128;

  v8f acc[4] = {};

  // ---------------- Phase 1: v_tile = x_tile @ T1 ----------------
  for (int nc = 0; nc < Nn; nc += 32) {
    const int c = nc >> 5;
    {
      // stage T1 fragments (compact global -> padded LDS), pure 16B copies
      const int t  = tid >> 6;
      const int l  = (tid >> 1) & 31;
      const int e0 = (tid & 1) << 3;
      const size_t g = ((size_t)(c * 4 + t) * 32 + l) * 16 + e0;
#if HAVE_ASYNC_LDS
      async_cp16(t1f_hi + g, &TfH[t][l * 24 + e0]);
      async_cp16(t1f_lo + g, &TfL[t][l * 24 + e0]);
#else
      *(v8h*)&TfH[t][l * 24 + e0] = *(const v8h*)(t1f_hi + g);
      *(v8h*)&TfL[t][l * 24 + e0] = *(const v8h*)(t1f_lo + g);
#endif
      // stage 128x32 of x -> hi/lo f16 (one half-row = 16 floats per thread)
      const int row = tid >> 1;
      const int c0  = (tid & 1) << 4;
      const v4f* xp = (const v4f*)(x + (m0 + row) * Nn + nc + c0);
      #pragma unroll
      for (int q = 0; q < 4; ++q) {
        const v4f f = __builtin_nontemporal_load(xp + q);  // one-pass stream
        v4h hi, lo;
        hi[0] = (_Float16)f[0]; lo[0] = (_Float16)(f[0] - (float)hi[0]);
        hi[1] = (_Float16)f[1]; lo[1] = (_Float16)(f[1] - (float)hi[1]);
        hi[2] = (_Float16)f[2]; lo[2] = (_Float16)(f[2] - (float)hi[2]);
        hi[3] = (_Float16)f[3]; lo[3] = (_Float16)(f[3] - (float)hi[3]);
        *(v4h*)&Xh[row][c0 + 4 * q] = hi;
        *(v4h*)&Xl[row][c0 + 4 * q] = lo;
      }
    }
    if (nc + 32 < Nn)  // global_prefetch_b8 for next chunk
      __builtin_prefetch(x + (m0 + (tid >> 1)) * Nn + nc + 32, 0, 0);
#if HAVE_ASYNC_LDS
    async_wait0();
#endif
    __syncthreads();

    const int mrow = wv * 16 + rr;
    v16h ah = frag2(&Xh[mrow][half * 16], &Xh[mrow][half * 16 + 8]);
    v16h al = frag2(&Xl[mrow][half * 16], &Xl[mrow][half * 16 + 8]);

    #pragma unroll
    for (int t = 0; t < 4; ++t) {
      v16h bh = frag2(&TfH[t][lane * 24], &TfH[t][lane * 24 + 8]);
      v16h bl = frag2(&TfL[t][lane * 24], &TfL[t][lane * 24 + 8]);
      acc[t] = WMMA_F16(ah, bh, acc[t]);  // hi*hi
      acc[t] = WMMA_F16(ah, bl, acc[t]);  // hi*lo
      acc[t] = WMMA_F16(al, bh, acc[t]);  // lo*hi
    }
    __syncthreads();
  }

  // write v (fp32 output #2, NT) and stash hi/lo split in LDS for phase 2
  #pragma unroll
  for (int t = 0; t < 4; ++t) {
    #pragma unroll
    for (int i = 0; i < 8; ++i) {
      const int row = wv * 16 + half * 8 + i;
      const int col = t * 16 + rr;
      const float val = acc[t][i];
      _Float16 h = (_Float16)val;
      Vh[row][col] = h;
      Vl[row][col] = (_Float16)(val - (float)h);
      __builtin_nontemporal_store(val, &vout[(m0 + row) * Kn + col]);
    }
  }
  __syncthreads();

  // ---------------- Phase 2: proj_tile = v_tile @ wn ----------------
  // A fragments (K = 64 -> two k-steps), loaded once, reused for all columns
  const int arow = wv * 16 + rr;
  v16h a0h = frag2(&Vh[arow][half * 16],      &Vh[arow][half * 16 + 8]);
  v16h a0l = frag2(&Vl[arow][half * 16],      &Vl[arow][half * 16 + 8]);
  v16h a1h = frag2(&Vh[arow][32 + half * 16], &Vh[arow][40 + half * 16]);
  v16h a1l = frag2(&Vl[arow][32 + half * 16], &Vl[arow][40 + half * 16]);

  for (int cc0 = 0; cc0 < Nn; cc0 += 128) {
    if (cc0) __syncthreads();  // previous chunk's fragment reads must finish
    {
      // stage 8 col-tiles x 2 k-steps of wn fragments, pure 16B copies
      const int ct0x2 = cc0 >> 4;
      #pragma unroll
      for (int q = 0; q < 4; ++q) {
        const int u  = q * 256 + tid;     // 0..1023 coalesced units of 8 halfs
        const int st = u >> 6;            // 0..15 : tile*2 + kstep
        const int l  = (u >> 1) & 31;
        const int e0 = (u & 1) << 3;
        const size_t g = ((size_t)(ct0x2 + st) * 32 + l) * 16 + e0;
#if HAVE_ASYNC_LDS
        async_cp16(wnf_hi + g, &WfH[st][l * 24 + e0]);
        async_cp16(wnf_lo + g, &WfL[st][l * 24 + e0]);
#else
        *(v8h*)&WfH[st][l * 24 + e0] = *(const v8h*)(wnf_hi + g);
        *(v8h*)&WfL[st][l * 24 + e0] = *(const v8h*)(wnf_lo + g);
#endif
      }
    }
#if HAVE_ASYNC_LDS
    async_wait0();
#endif
    __syncthreads();

    #pragma unroll
    for (int t = 0; t < 8; ++t) {
      v16h b0h = frag2(&WfH[t * 2][lane * 24],     &WfH[t * 2][lane * 24 + 8]);
      v16h b0l = frag2(&WfL[t * 2][lane * 24],     &WfL[t * 2][lane * 24 + 8]);
      v16h b1h = frag2(&WfH[t * 2 + 1][lane * 24], &WfH[t * 2 + 1][lane * 24 + 8]);
      v16h b1l = frag2(&WfL[t * 2 + 1][lane * 24], &WfL[t * 2 + 1][lane * 24 + 8]);
      v8f cacc = {};
      cacc = WMMA_F16(a0h, b0h, cacc);
      cacc = WMMA_F16(a0h, b0l, cacc);
      cacc = WMMA_F16(a0l, b0h, cacc);
      cacc = WMMA_F16(a1h, b1h, cacc);
      cacc = WMMA_F16(a1h, b1l, cacc);
      cacc = WMMA_F16(a1l, b1h, cacc);
      #pragma unroll
      for (int i = 0; i < 8; ++i) {
        const int row = wv * 16 + half * 8 + i;
        __builtin_nontemporal_store(cacc[i],
            &proj[(m0 + row) * Nn + cc0 + t * 16 + rr]);  // write-once stream
      }
    }
  }
}

// ---------------------------------------------------------------------------
extern "C" void kernel_launch(void* const* d_in, const int* in_sizes, int n_in,
                              void* d_out, int out_size, void* d_ws, size_t ws_size,
                              hipStream_t stream) {
  const float* x = (const float*)d_in[0];   // (B, N)
  const float* w = (const float*)d_in[1];   // (K, N)
  const float* r = (const float*)d_in[2];   // (K, K)

  float* proj = (float*)d_out;              // (B, N)
  float* vout = proj + (size_t)Bn * Nn;     // (B, K)

  // workspace: wn32 (256KB) | t1f_hi | t1f_lo | wnf_hi | wnf_lo (128KB each)
  char* ws = (char*)d_ws;
  float*    wn32   = (float*)ws;
  _Float16* t1f_hi = (_Float16*)(ws + (size_t)Kn * Nn * sizeof(float));
  _Float16* t1f_lo = t1f_hi + (size_t)Nn * Kn;
  _Float16* wnf_hi = t1f_lo + (size_t)Nn * Kn;
  _Float16* wnf_lo = wnf_hi + (size_t)Kn * Nn;

  spaceopt_prep<<<1, 256, 0, stream>>>(w, r, wn32, t1f_hi, t1f_lo, wnf_hi, wnf_lo);
  spaceopt_main<<<Bn / 128, 256, 0, stream>>>(x, t1f_hi, t1f_lo, wnf_hi, wnf_lo,
                                              proj, vout);
}
